// LSTM_57251914056386
// MI455X (gfx1250) — compile-verified, matrix operand order checked by reading, not run on hardware
//
#include <hip/hip_runtime.h>
#include <cmath>

// ---------------- problem sizes ----------------
constexpr int B  = 128;
constexpr int T  = 64;
constexpr int H  = 4096;
constexpr int E  = 256;
constexpr int V  = 256;
constexpr int G  = 4 * H;        // 16384
constexpr int BT = B * T;        // 8192

// ---------------- vector types ----------------
typedef __attribute__((ext_vector_type(16))) __bf16 v16bf;
typedef __attribute__((ext_vector_type(8)))  __bf16 v8bf;
typedef __attribute__((ext_vector_type(8)))  float  v8f;

// float -> bf16 (round-to-nearest-even)
static __device__ __forceinline__ __bf16 f2bf(float f) {
  union { float f; unsigned u; } a; a.f = f;
  unsigned r = a.u + 0x7FFFu + ((a.u >> 16) & 1u);
  unsigned short h = (unsigned short)(r >> 16);
  return __builtin_bit_cast(__bf16, h);
}

static __device__ __forceinline__ float sigmoidf(float x) {
  return 1.0f / (1.0f + __expf(-x));
}

// Load a 16x32 (A) / 32x16 (B) bf16 WMMA fragment for one lane.
// ISA 7.12.2: lane (l&15) owns the row/col; lane>>4 selects the K-half.
// Per lane the 16 elements are two contiguous 8-elem runs:
//   K in [lh*8, lh*8+8) and [16+lh*8, 16+lh*8+8)   -> two global_load_b128
static __device__ __forceinline__ v16bf load_frag(const __bf16* __restrict__ rowp,
                                                  int kk, int lh) {
  v8bf lo = *(const v8bf*)(rowp + kk + lh * 8);
  v8bf hi = *(const v8bf*)(rowp + kk + 16 + lh * 8);
  v16bf f;
#pragma unroll
  for (int i = 0; i < 8; ++i) { f[i] = lo[i]; f[8 + i] = hi[i]; }
  return f;
}

#define WMMA_BF16(Afrag, Bfrag, Cacc) \
  __builtin_amdgcn_wmma_f32_16x16x32_bf16(false, (Afrag), false, (Bfrag), (short)0, (Cacc), false, false)

// ---------------------------------------------------------------------------
// 1) weight_norm rows + quantize to bf16.  scale = g[row] / ||v_row||  (g==null -> 1)
// ---------------------------------------------------------------------------
__global__ __launch_bounds__(256)
void rownorm_bf16_kernel(const float* __restrict__ v, const float* __restrict__ g,
                         __bf16* __restrict__ out, int C) {
  __shared__ float red[256];
  const int row = blockIdx.x;
  const float* vr = v + (size_t)row * C;
  float s = 0.f;
  for (int c = threadIdx.x; c < C; c += 256) { float x = vr[c]; s += x * x; }
  red[threadIdx.x] = s;
  __syncthreads();
  for (int off = 128; off > 0; off >>= 1) {
    if ((int)threadIdx.x < off) red[threadIdx.x] += red[threadIdx.x + off];
    __syncthreads();
  }
  float scale = 1.0f;
  if (g) scale = g[row] * rsqrtf(red[0]);
  for (int c = threadIdx.x; c < C; c += 256)
    out[(size_t)row * C + c] = f2bf(vr[c] * scale);
}

// ---------------------------------------------------------------------------
// 2) embedding gather -> bf16   xe[m, e] = emb[x[m], e]
// ---------------------------------------------------------------------------
__global__ __launch_bounds__(256)
void gather_embed_kernel(const int* __restrict__ x, const float* __restrict__ emb,
                         __bf16* __restrict__ xe) {
  const int m = blockIdx.x;
  const int tok = x[m];
  const float* er = emb + (size_t)tok * E;
  for (int e = threadIdx.x; e < E; e += 256)
    xe[(size_t)m * E + e] = f2bf(er[e]);
}

// ---------------------------------------------------------------------------
// 3) xg = xe @ W_ih^T + b_ih + b_hh   [BT, G] fp32
//    grid (BT/16, G/128), block 256 = 8 waves, each wave one 16x16 tile
// ---------------------------------------------------------------------------
__global__ __launch_bounds__(256)
void xg_gemm_kernel(const __bf16* __restrict__ xe, const __bf16* __restrict__ wih,
                    const float* __restrict__ b_ih, const float* __restrict__ b_hh,
                    float* __restrict__ xg) {
  const int wave = threadIdx.x >> 5, lane = threadIdx.x & 31;
  const int lr = lane & 15, lh = lane >> 4;
  const int m0 = blockIdx.x * 16;
  const int n0 = blockIdx.y * 128 + wave * 16;

  v8f acc = {};
  const __bf16* arow = xe  + (size_t)(m0 + lr) * E;
  const __bf16* brow = wih + (size_t)(n0 + lr) * E;
#pragma unroll
  for (int kk = 0; kk < E; kk += 32) {
    v16bf a = load_frag(arow, kk, lh);
    v16bf b = load_frag(brow, kk, lh);
    acc = WMMA_BF16(a, b, acc);
  }
  const int n = n0 + lr;
  const float bias = b_ih[n] + b_hh[n];
#pragma unroll
  for (int r = 0; r < 8; ++r) {
    const int m = m0 + lh * 8 + r;           // C/D layout: m = r + 8*(lane>>4)
    xg[(size_t)m * G + n] = acc[r] + bias;
  }
}

// ---------------------------------------------------------------------------
// 4) init recurrent state
// ---------------------------------------------------------------------------
__global__ __launch_bounds__(256)
void init_state_kernel(const float* __restrict__ h0, const float* __restrict__ c0,
                       __bf16* __restrict__ hbf, float* __restrict__ cws) {
  const size_t i = (size_t)blockIdx.x * 256 + threadIdx.x;
  if (i < (size_t)B * H) { hbf[i] = f2bf(h0[i]); cws[i] = c0[i]; }
}

// ---------------------------------------------------------------------------
// 5) fused LSTM step: gates = xg[:,t,:] + h_prev @ W_hh^T  -> pointwise -> h,c
//    grid = H/32 = 128 blocks; block = 256 thr = 8 waves covering all B=128 rows.
//    Each wave: 16(batch) x 32(h-col, two 16-wide subtiles) x 4 gates
//    -> 8 f32 accumulators; one A fragment feeds 8 WMMAs (A reuse x8).
//    W_hh is bf16 (128 MB -> L2-resident on MI455X's 192 MB L2), so each of
//    the 64 sequential steps streams weights from L2, not HBM.
// ---------------------------------------------------------------------------
__global__ __launch_bounds__(256)
void lstm_step_kernel(const __bf16* __restrict__ hprev, const __bf16* __restrict__ whh,
                      const float* __restrict__ xg, float* __restrict__ cws, int t,
                      float* __restrict__ all_h, __bf16* __restrict__ hnext,
                      __bf16* __restrict__ hall, float* __restrict__ hT,
                      float* __restrict__ cT) {
  const int wave = threadIdx.x >> 5, lane = threadIdx.x & 31;
  const int lr = lane & 15, lh = lane >> 4;
  const int m0 = wave * 16;              // batch rows
  const int n0 = blockIdx.x * 32;        // hidden column base (two subtiles)

  v8f acc[4][2];
#pragma unroll
  for (int g = 0; g < 4; ++g)
#pragma unroll
    for (int s = 0; s < 2; ++s) acc[g][s] = (v8f){};

  const __bf16* arow = hprev + (size_t)(m0 + lr) * H;
  const __bf16* brow[4][2];
#pragma unroll
  for (int g = 0; g < 4; ++g)
#pragma unroll
    for (int s = 0; s < 2; ++s)
      brow[g][s] = whh + ((size_t)g * H + n0 + s * 16 + lr) * (size_t)H;

#pragma unroll 2
  for (int kk = 0; kk < H; kk += 32) {
    // hint upcoming K-chunks of the streaming B operand into cache
    __builtin_prefetch(brow[0][0] + kk + 128, 0, 1);
    __builtin_prefetch(brow[2][0] + kk + 128, 0, 1);
    v16bf a = load_frag(arow, kk, lh);
#pragma unroll
    for (int g = 0; g < 4; ++g) {
#pragma unroll
      for (int s = 0; s < 2; ++s) {
        v16bf b = load_frag(brow[g][s], kk, lh);
        acc[g][s] = WMMA_BF16(a, b, acc[g][s]);
      }
    }
  }

#pragma unroll
  for (int s = 0; s < 2; ++s) {
    const int n = n0 + s * 16 + lr;
#pragma unroll
    for (int r = 0; r < 8; ++r) {
      const int m = m0 + lh * 8 + r;                       // batch index
      const size_t xgb = ((size_t)m * T + t) * (size_t)G;
      const float iv = acc[0][s][r] + xg[xgb + 0 * (size_t)H + n];
      const float fv = acc[1][s][r] + xg[xgb + 1 * (size_t)H + n];
      const float gv = acc[2][s][r] + xg[xgb + 2 * (size_t)H + n];
      const float ov = acc[3][s][r] + xg[xgb + 3 * (size_t)H + n];

      const float cp = cws[(size_t)m * H + n];
      const float cn = sigmoidf(fv) * cp + sigmoidf(iv) * tanhf(gv);
      const float hn = sigmoidf(ov) * tanhf(cn);

      cws[(size_t)m * H + n] = cn;
      all_h[((size_t)m * T + t) * (size_t)H + n] = hn;
      const __bf16 hb = f2bf(hn);
      hnext[(size_t)m * H + n] = hb;
      hall[((size_t)m * T + t) * (size_t)H + n] = hb;
      if (t == T - 1) {
        hT[(size_t)m * H + n] = hn;
        cT[(size_t)m * H + n] = cn;
      }
    }
  }
}

// ---------------------------------------------------------------------------
// 6) logits = all_h @ w_out^T + b_out   [BT, V]
//    grid (BT/128, V/16), block 256 = 8 waves (wave -> m-subtile)
// ---------------------------------------------------------------------------
__global__ __launch_bounds__(256)
void logits_gemm_kernel(const __bf16* __restrict__ hall, const __bf16* __restrict__ wout,
                        const float* __restrict__ b_out, float* __restrict__ logits) {
  const int wave = threadIdx.x >> 5, lane = threadIdx.x & 31;
  const int lr = lane & 15, lh = lane >> 4;
  const int m0 = blockIdx.x * 128 + wave * 16;
  const int n0 = blockIdx.y * 16;

  v8f acc = {};
  const __bf16* arow = hall + (size_t)(m0 + lr) * H;
  const __bf16* brow = wout + (size_t)(n0 + lr) * H;
#pragma unroll 2
  for (int kk = 0; kk < H; kk += 32) {
    v16bf a = load_frag(arow, kk, lh);
    v16bf b = load_frag(brow, kk, lh);
    acc = WMMA_BF16(a, b, acc);
  }
  const int nn = n0 + lr;
  const float bias = b_out[nn];
#pragma unroll
  for (int r = 0; r < 8; ++r) {
    const int m = m0 + lh * 8 + r;
    logits[(size_t)m * V + nn] = acc[r] + bias;
  }
}

// ---------------------------------------------------------------------------
// host-side orchestration
// ---------------------------------------------------------------------------
extern "C" void kernel_launch(void* const* d_in, const int* in_sizes, int n_in,
                              void* d_out, int out_size, void* d_ws, size_t ws_size,
                              hipStream_t stream) {
  (void)in_sizes; (void)n_in; (void)out_size; (void)ws_size;

  // inputs (setup_inputs order)
  const int*   x     = (const int*)  d_in[0];
  const float* h0    = (const float*)d_in[1];
  const float* c0    = (const float*)d_in[2];
  const float* emb   = (const float*)d_in[3];
  const float* v_ih  = (const float*)d_in[4];
  const float* g_ih  = (const float*)d_in[5];
  const float* v_hh  = (const float*)d_in[6];
  const float* g_hh  = (const float*)d_in[7];
  const float* b_ih  = (const float*)d_in[8];
  const float* b_hh  = (const float*)d_in[9];
  const float* w_out = (const float*)d_in[10];
  const float* b_out = (const float*)d_in[11];

  // outputs (tuple order: logits, hT, cT, all_h)
  float* out    = (float*)d_out;
  float* logits = out;
  float* hT     = logits + (size_t)BT * V;
  float* cT     = hT + (size_t)B * H;
  float* all_h  = cT + (size_t)B * H;

  // workspace layout (bytes)
  char* ws = (char*)d_ws;
  size_t off = 0;
  auto take = [&](size_t bytes) { char* p = ws + off; off += (bytes + 255) & ~(size_t)255; return p; };
  __bf16* wih_bf  = (__bf16*)take((size_t)G * E * 2);   //   8 MB
  __bf16* whh_bf  = (__bf16*)take((size_t)G * H * 2);   // 128 MB (fits MI455X L2)
  __bf16* wout_bf = (__bf16*)take((size_t)V * H * 2);   //   2 MB
  __bf16* xe_bf   = (__bf16*)take((size_t)BT * E * 2);  //   4 MB
  float*  xg      = (float*) take((size_t)BT * G * 4);  // 512 MB
  __bf16* hbf0    = (__bf16*)take((size_t)B * H * 2);   //   1 MB
  __bf16* hbf1    = (__bf16*)take((size_t)B * H * 2);   //   1 MB
  float*  cws     = (float*) take((size_t)B * H * 4);   //   2 MB
  __bf16* hall_bf = (__bf16*)take((size_t)BT * H * 2);  //  64 MB

  // 1) weight-norm + bf16 quantize
  rownorm_bf16_kernel<<<G, 256, 0, stream>>>(v_ih, g_ih, wih_bf, E);
  rownorm_bf16_kernel<<<G, 256, 0, stream>>>(v_hh, g_hh, whh_bf, H);
  rownorm_bf16_kernel<<<V, 256, 0, stream>>>(w_out, nullptr, wout_bf, H);

  // 2) embedding gather
  gather_embed_kernel<<<BT, 256, 0, stream>>>(x, emb, xe_bf);

  // 3) input projection (both biases folded in)
  xg_gemm_kernel<<<dim3(BT / 16, G / 128), 256, 0, stream>>>(xe_bf, wih_bf, b_ih, b_hh, xg);

  // 4) state init
  init_state_kernel<<<((size_t)B * H + 255) / 256, 256, 0, stream>>>(h0, c0, hbf0, cws);

  // 5) recurrence: 64 sequential fused step launches, ping-pong h buffers
  for (int t = 0; t < T; ++t) {
    const __bf16* hprev = (t & 1) ? hbf1 : hbf0;
    __bf16*       hnext = (t & 1) ? hbf0 : hbf1;
    lstm_step_kernel<<<H / 32, 256, 0, stream>>>(hprev, whh_bf, xg, cws, t,
                                                 all_h, hnext, hall_bf, hT, cT);
  }

  // 6) output projection
  logits_gemm_kernel<<<dim3(BT / 128, V / 16), 256, 0, stream>>>(hall_bf, wout_bf, b_out, logits);
}
